// Attention_30374008717731
// MI455X (gfx1250) — matrix operand order repaired and checked by reading, not compiled, not run on hardware
//
#include <hip/hip_runtime.h>
#include <hip/hip_bf16.h>
#include <math.h>

// Shapes fixed by the reference
#define LL 128
#define BB 128
#define DD 1024
#define AA 1024
#define HH 2048

typedef __attribute__((ext_vector_type(16))) __bf16 v16bf;
typedef __attribute__((ext_vector_type(8)))  float  v8f;

// Fast tanh: 1 - 2/(exp(2x)+1).  v_exp_f32 + v_rcp_f32, saturates to +/-1.
__device__ __forceinline__ float fast_tanh(float x) {
    float e = __expf(2.0f * x);
    return 1.0f - 2.0f * __builtin_amdgcn_rcpf(e + 1.0f);
}

// ---------------------------------------------------------------------------
// Kernel 1: proj[b][a] = sum_d s_tm1[b][d] * W_sa[a][d] + b_sa[a]
// WMMA bf16 16x16x32, one wave (32 lanes) per 16x16 output tile.
// M = B = 128 (8 tiles), N = A = 1024 (64 tiles), K = D = 1024 (32 steps).
// ---------------------------------------------------------------------------
__global__ void __launch_bounds__(256)
proj_wmma_kernel(const float* __restrict__ s_tm1,
                 const float* __restrict__ W_sa,
                 const float* __restrict__ b_sa,
                 float* __restrict__ proj)
{
    const int lane = threadIdx.x & 31;
    const int wave = threadIdx.x >> 5;
    const int tile = blockIdx.x * 8 + wave;       // 512 tiles total
    const int tm   = tile >> 6;                   // 0..7   (M tiles)
    const int tn   = tile & 63;                   // 0..63  (N tiles)
    const int m0   = tm * 16;
    const int n0   = tn * 16;

    const int m    = lane & 15;                   // row within tile (A-matrix)
    const int n    = lane & 15;                   // col within tile (B/C/D)
    const int half = lane >> 4;                   // lane group 0/1

    v8f c = {};
    for (int kb = 0; kb < DD; kb += 32) {
        // A-matrix (16x32 bf16, contiguous k = 16*half + s, same as B): lane group selects K sub-ranges.
        const float* arow = s_tm1 + (size_t)(m0 + m) * DD + kb + 16 * half;
        v16bf a, b;
#pragma unroll
        for (int j = 0; j < 8; ++j) {
            a[j]     = (__bf16)arow[j];
            a[j + 8] = (__bf16)arow[8 + j];
        }
        // B-matrix (32x16 bf16): lanes 0-15 hold K=kb..kb+15 for col n,
        // lanes 16-31 hold K=kb+16..kb+31.  B[k][n] = W_sa[n][k].
        const float* brow = W_sa + (size_t)(n0 + n) * DD + kb + 16 * half;
#pragma unroll
        for (int j = 0; j < 16; ++j) b[j] = (__bf16)brow[j];

        c = __builtin_amdgcn_wmma_f32_16x16x32_bf16(
                /*neg_a=*/false, a, /*neg_b=*/false, b,
                /*c_mod=*/(short)0, c, /*reuse_a=*/false, /*reuse_b=*/false);
    }

    const float bias = b_sa[n0 + n];
    // C/D layout: VGPR r, lanes 0-15 -> M = r, lanes 16-31 -> M = r + 8
#pragma unroll
    for (int r = 0; r < 8; ++r) {
        proj[(size_t)(m0 + r + 8 * half) * AA + n0 + n] = c[r] + bias;
    }
}

// ---------------------------------------------------------------------------
// Kernel 2: e_raw[l][b] = exp( sum_a tanh(proj[b][a] + uh[l][b][a]) * w_a1[a]
//                              + b_a1 ) * mask[l][b]
// One wave per (l,b); float4 (b128) streaming of uh; wave32 xor-reduction.
// ---------------------------------------------------------------------------
__global__ void __launch_bounds__(256)
score_kernel(const float* __restrict__ proj,
             const float* __restrict__ uh,
             const float* __restrict__ w_a1,
             const float* __restrict__ b_a1,
             const float* __restrict__ xs_mask,
             float* __restrict__ e_raw)
{
    const int lane = threadIdx.x & 31;
    const int wave = threadIdx.x >> 5;
    const int idx  = blockIdx.x * 8 + wave;       // l*B + b, 16384 total
    const int b    = idx & (BB - 1);

    const float4* up = (const float4*)(uh   + (size_t)idx * AA);
    const float4* pp = (const float4*)(proj + (size_t)b   * AA);
    const float4* wp = (const float4*)w_a1;

    float s = 0.0f;
#pragma unroll 2
    for (int a4 = lane; a4 < AA / 4; a4 += 32) {
        const float4 u = up[a4];
        const float4 p = pp[a4];
        const float4 w = wp[a4];
        s = fmaf(fast_tanh(p.x + u.x), w.x, s);
        s = fmaf(fast_tanh(p.y + u.y), w.y, s);
        s = fmaf(fast_tanh(p.z + u.z), w.z, s);
        s = fmaf(fast_tanh(p.w + u.w), w.w, s);
    }

    // wave32 reduction
#pragma unroll
    for (int off = 16; off > 0; off >>= 1)
        s += __shfl_xor(s, off, 32);

    if (lane == 0)
        e_raw[idx] = expf(s + b_a1[0]) * xs_mask[idx];
}

// ---------------------------------------------------------------------------
// Kernel 3: normalize over L per batch column b; write e_ij to d_out[0..L*B)
// ---------------------------------------------------------------------------
__global__ void __launch_bounds__(128)
normalize_kernel(const float* __restrict__ e_raw,
                 float* __restrict__ e_out)
{
    __shared__ float red[LL];
    const int b = blockIdx.x;
    const int l = threadIdx.x;                    // 128 threads = L

    const float v = e_raw[l * BB + b];
    red[l] = v;
    __syncthreads();
#pragma unroll
    for (int s = LL / 2; s > 0; s >>= 1) {
        if (l < s) red[l] += red[l + s];
        __syncthreads();
    }
    e_out[l * BB + b] = v * (1.0f / red[0]);
}

// ---------------------------------------------------------------------------
// Kernel 4: attend[b][h] = sum_l e[l][b] * xs_h[l][b][h]
// Block = (b, 1024-wide h stripe, float4 per thread); e column in LDS.
// Streams the dominant 128 MB of xs_h with b128 loads.
// ---------------------------------------------------------------------------
__global__ void __launch_bounds__(256)
attend_kernel(const float* __restrict__ e,      // normalized, in d_out[0..L*B)
              const float* __restrict__ xs_h,
              float* __restrict__ out_attend)   // d_out + L*B
{
    __shared__ float se[LL];
    const int b  = blockIdx.y;
    const int h4 = blockIdx.x * 256 + threadIdx.x;   // float4 index within H

    if (threadIdx.x < LL) se[threadIdx.x] = e[threadIdx.x * BB + b];
    __syncthreads();

    const float4* xp = (const float4*)(xs_h + (size_t)b * HH) + h4;
    const size_t  lstride4 = (size_t)BB * HH / 4;    // float4 stride per l

    float4 acc = {0.0f, 0.0f, 0.0f, 0.0f};
#pragma unroll 4
    for (int l = 0; l < LL; ++l) {
        if (l + 8 < LL)
            __builtin_prefetch(xp + (size_t)(l + 8) * lstride4, 0, 1);
        const float4 x = xp[(size_t)l * lstride4];
        const float  w = se[l];
        acc.x = fmaf(w, x.x, acc.x);
        acc.y = fmaf(w, x.y, acc.y);
        acc.z = fmaf(w, x.z, acc.z);
        acc.w = fmaf(w, x.w, acc.w);
    }
    ((float4*)(out_attend + (size_t)b * HH))[h4] = acc;
}

// ---------------------------------------------------------------------------
extern "C" void kernel_launch(void* const* d_in, const int* in_sizes, int n_in,
                              void* d_out, int out_size, void* d_ws, size_t ws_size,
                              hipStream_t stream)
{
    const float* s_tm1   = (const float*)d_in[0];   // (B, D)
    const float* xs_h    = (const float*)d_in[1];   // (L, B, H)
    const float* uh      = (const float*)d_in[2];   // (L, B, A)
    const float* xs_mask = (const float*)d_in[3];   // (L, B)
    const float* W_sa    = (const float*)d_in[4];   // (A, D)
    const float* b_sa    = (const float*)d_in[5];   // (A,)
    const float* w_a1    = (const float*)d_in[6];   // (A,)
    const float* b_a1    = (const float*)d_in[7];   // scalar

    float* out   = (float*)d_out;                   // [e_ij (L*B) | attend (B*H)]
    float* proj  = (float*)d_ws;                    // B*A floats
    float* e_raw = proj + (size_t)BB * AA;          // L*B floats

    // 1) proj = s_tm1 @ W_sa^T + b_sa   (WMMA bf16, 512 tiles / 8 waves per block)
    proj_wmma_kernel<<<64, 256, 0, stream>>>(s_tm1, W_sa, b_sa, proj);

    // 2) raw attention scores (one wave per (l,b), b128 streaming of uh)
    score_kernel<<<(LL * BB) / 8, 256, 0, stream>>>(proj, uh, w_a1, b_a1,
                                                    xs_mask, e_raw);

    // 3) normalize over L; writes e_ij into d_out[0 .. L*B)
    normalize_kernel<<<BB, LL, 0, stream>>>(e_raw, out);

    // 4) attend = einsum('lb,lbh->bh'); writes d_out[L*B ..)
    attend_kernel<<<dim3(HH / 1024, BB), 256, 0, stream>>>(out, xs_h,
                                                           out + LL * BB);
}
